// SparseMatrixEmbed_WACO_NET_10952166605483
// MI455X (gfx1250) — compile-verified, matrix operand order checked by reading, not run on hardware
//
#include <hip/hip_runtime.h>

// ---------- types for WMMA bf16 path ----------
typedef __attribute__((ext_vector_type(16))) __bf16        v16bf;
typedef __attribute__((ext_vector_type(16))) unsigned short v16us;
typedef __attribute__((ext_vector_type(8)))  unsigned int   v8ui;
typedef __attribute__((ext_vector_type(8)))  float          v8f;

union Frag {
    v16us u16;
    v8ui  u32;
    v16bf bf;
};

__device__ __forceinline__ unsigned short f32_to_bf16(float f) {
    union { float f; unsigned int u; } x; x.f = f;
    unsigned int r = x.u + 0x7FFFu + ((x.u >> 16) & 1u);   // round-to-nearest-even
    return (unsigned short)(r >> 16);
}
__device__ __forceinline__ float bf16_to_f32(unsigned short h) {
    union { unsigned int u; float f; } x; x.u = ((unsigned int)h) << 16;
    return x.f;
}
__device__ __forceinline__ unsigned int pack_bf16x2(float lo, float hi) {
    return (unsigned int)f32_to_bf16(lo) | ((unsigned int)f32_to_bf16(hi) << 16);
}
__device__ __forceinline__ int iclamp(int v, int lo, int hi) {
    v = v < lo ? lo : v;
    return v > hi ? hi : v;
}

#define LDS_ROW 20   // floats per pixel row in transpose tile (80B pad, 16B-aligned)

// Activation maps are stored PADDED: (H+2)x(W+2)x32 bf16 with a zero 1-pixel
// halo (3x3 SAME without bounds checks).  The raw input x gets a bf16 padded
// copy with a 2-pixel halo for the 5x5 layer.

// ---------------------------------------------------------------------------
// Input prep: x (H*W fp32) -> xpad ((H+4)*(W+4) bf16, zero 2-pixel halo).
// ---------------------------------------------------------------------------
__global__ void pad_input(const float* __restrict__ x,
                          unsigned short* __restrict__ xpad,
                          int H, int W)
{
    const int Wq = W + 4;
    const int nP = (H + 4) * Wq;
    for (int i = blockIdx.x * blockDim.x + threadIdx.x; i < nP;
         i += gridDim.x * blockDim.x) {
        int py = i / Wq, px = i % Wq;
        int sy = py - 2, sx = px - 2;
        bool inb = sy >= 0 && sy < H && sx >= 0 && sx < W;
        int syc = iclamp(sy, 0, H - 1), sxc = iclamp(sx, 0, W - 1);
        float v = x[(size_t)syc * W + sxc];          // unconditional, clamped
        float flag = inb ? 1.f : 0.f;
        xpad[i] = f32_to_bf16(v * flag);
    }
}

// ---------------------------------------------------------------------------
// Layer 1: 5x5 conv, C_in=1 -> C_out=32, stride 1, SAME, relu * mask.
// K = 25 taps padded to 32 -> one v_wmma_f32_16x16x32_bf16 per 16-pixel tile.
// A gather = 16 plain bf16 loads from the padded copy (no guards/converts);
// pad slots clamp to tap 24 and are killed by the zero B columns.
// ---------------------------------------------------------------------------
__global__ void conv5x5_wmma(const unsigned short* __restrict__ xpad, // (H+4)*(W+4) bf16
                             unsigned short* __restrict__ out,        // (H+2)*(W+2)*32 bf16
                             const float* __restrict__ mask,          // H*W
                             const unsigned short* __restrict__ Bw0,  // 32*32 bf16 [k][n]
                             int H, int W, int lgW)
{
    __shared__ float xpose[4][16 * LDS_ROW];

    const int lane  = threadIdx.x & 31;
    const int wave  = threadIdx.x >> 5;
    const int wpb   = blockDim.x >> 5;
    const int gw    = blockIdx.x * wpb + wave;
    const int nHalf = gw & 1;
    const int pairIdx = gw >> 1;
    const int pairStride = (gridDim.x * wpb) >> 1;

    const int m  = lane & 15;
    const int hi = lane >> 4;
    const int n  = lane & 15;
    const int nBase = nHalf * 16;
    const int Wp = W + 2;
    const int Wq = W + 4;
    float* lds = xpose[wave];

    Frag B;
    {
        v16us bu;
#pragma unroll
        for (int e = 0; e < 16; ++e) {
            int k = ((e & 8) << 1) + hi * 8 + (e & 7);
            bu[e] = Bw0[k * 32 + nBase + n];
        }
        B.u16 = bu;
    }

    const int nPix = H * W;
    const int nTiles = (nPix + 15) >> 4;

    for (int tile = pairIdx; tile < nTiles; tile += pairStride) {
        const int pixRaw = tile * 16 + m;
        const int pix = pixRaw < nPix ? pixRaw : nPix - 1;
        const int py = pix >> lgW;
        const int px = pix & (W - 1);
        const unsigned short* srcBase = xpad + (size_t)py * Wq + px;  // tap (-2,-2) is +0

        Frag A;
#pragma unroll
        for (int e = 0; e < 16; ++e) {
            int k = ((e & 8) << 1) + hi * 8 + (e & 7);   // tap index; 25..31 pad
            int kk = k < 25 ? k : 24;                    // valid dup load, B col is 0
            int dy = kk / 5, dx = kk % 5;                // 0..4 in padded coords
            A.u16[e] = srcBase[(size_t)dy * Wq + dx];
        }

        v8f acc = {0.f, 0.f, 0.f, 0.f, 0.f, 0.f, 0.f, 0.f};
        acc = __builtin_amdgcn_wmma_f32_16x16x32_bf16(false, A.bf, false, B.bf,
                                                      (short)0, acc, false, false);

        // ---- LDS transpose epilogue (wave-private tile) ----
#pragma unroll
        for (int r = 0; r < 8; ++r)
            lds[(r + 8 * hi) * LDS_ROW + n] = acc[r];
        asm volatile("s_wait_dscnt 0x0" ::: "memory");

        const int j  = lane >> 1;
        const int c8 = (lane & 1) * 8;
        const int p  = tile * 16 + j;
        const int pc = p < nPix ? p : nPix - 1;
        const int pyk = pc >> lgW, pxk = pc & (W - 1);
        const float mk = mask[pc];

        float v0 = lds[j * LDS_ROW + c8 + 0], v1 = lds[j * LDS_ROW + c8 + 1];
        float v2 = lds[j * LDS_ROW + c8 + 2], v3 = lds[j * LDS_ROW + c8 + 3];
        float v4 = lds[j * LDS_ROW + c8 + 4], v5 = lds[j * LDS_ROW + c8 + 5];
        float v6 = lds[j * LDS_ROW + c8 + 6], v7 = lds[j * LDS_ROW + c8 + 7];
        v0 = fmaxf(v0, 0.f) * mk; v1 = fmaxf(v1, 0.f) * mk;
        v2 = fmaxf(v2, 0.f) * mk; v3 = fmaxf(v3, 0.f) * mk;
        v4 = fmaxf(v4, 0.f) * mk; v5 = fmaxf(v5, 0.f) * mk;
        v6 = fmaxf(v6, 0.f) * mk; v7 = fmaxf(v7, 0.f) * mk;
        uint4 st;
        st.x = pack_bf16x2(v0, v1); st.y = pack_bf16x2(v2, v3);
        st.z = pack_bf16x2(v4, v5); st.w = pack_bf16x2(v6, v7);
        if (p < nPix)
            *(uint4*)(out + ((size_t)(pyk + 1) * Wp + (pxk + 1)) * 32 + nBase + c8) = st;
    }
}

// ---------------------------------------------------------------------------
// 3x3 conv, 32->32, stride 2, SAME (via padded halo), relu * maskOut.
// Inner loop per tap: address add + two global b128 loads + one WMMA.
// ---------------------------------------------------------------------------
__global__ void conv3x3s2_wmma(const unsigned short* __restrict__ in,   // (Hin+2)*(Win+2)*32
                               unsigned short* __restrict__ out,        // (Hout+2)*(Wout+2)*32
                               const float* __restrict__ maskOut,       // Hout*Wout
                               const unsigned short* __restrict__ Bw,   // 288*32 bf16 [k][n]
                               int Win, int Hout, int Wout, int lgWout)
{
    __shared__ float xpose[4][16 * LDS_ROW];

    const int lane  = threadIdx.x & 31;
    const int wave  = threadIdx.x >> 5;
    const int wpb   = blockDim.x >> 5;
    const int gw    = blockIdx.x * wpb + wave;
    const int nHalf = gw & 1;
    const int pairIdx = gw >> 1;
    const int pairStride = (gridDim.x * wpb) >> 1;

    const int m  = lane & 15;
    const int hi = lane >> 4;
    const int n  = lane & 15;
    const int nBase = nHalf * 16;
    const int Winp = Win + 2;
    const int Woutp = Wout + 2;
    float* lds = xpose[wave];

    Frag B[9];
#pragma unroll
    for (int t = 0; t < 9; ++t) {
        v16us bu;
#pragma unroll
        for (int e = 0; e < 16; ++e) {
            int k = ((e & 8) << 1) + hi * 8 + (e & 7);       // c_in within this tap
            bu[e] = Bw[(t * 32 + k) * 32 + nBase + n];
        }
        B[t].u16 = bu;
    }

    const int nPix = Hout * Wout;
    const int nTiles = (nPix + 15) >> 4;

    for (int tile = pairIdx; tile < nTiles; tile += pairStride) {
        const int pixRaw = tile * 16 + m;
        const int pix = pixRaw < nPix ? pixRaw : nPix - 1;
        const int oy = pix >> lgWout;
        const int ox = pix & (Wout - 1);
        const unsigned short* srcBase =
            in + ((size_t)(oy * 2) * Winp + (ox * 2)) * 32 + hi * 8;

        v8f acc = {0.f, 0.f, 0.f, 0.f, 0.f, 0.f, 0.f, 0.f};
#pragma unroll
        for (int t = 0; t < 9; ++t) {
            const unsigned short* src = srcBase + ((size_t)(t / 3) * Winp + (t % 3)) * 32;
            uint4 c0 = *(const uint4*)(src);
            uint4 c1 = *(const uint4*)(src + 16);
            Frag A;
            A.u32[0] = c0.x; A.u32[1] = c0.y; A.u32[2] = c0.z; A.u32[3] = c0.w;
            A.u32[4] = c1.x; A.u32[5] = c1.y; A.u32[6] = c1.z; A.u32[7] = c1.w;
            acc = __builtin_amdgcn_wmma_f32_16x16x32_bf16(false, A.bf, false, B[t].bf,
                                                          (short)0, acc, false, false);
        }

        // ---- LDS transpose epilogue (wave-private tile) ----
#pragma unroll
        for (int r = 0; r < 8; ++r)
            lds[(r + 8 * hi) * LDS_ROW + n] = acc[r];
        asm volatile("s_wait_dscnt 0x0" ::: "memory");

        const int j  = lane >> 1;
        const int c8 = (lane & 1) * 8;
        const int p  = tile * 16 + j;
        const int pc = p < nPix ? p : nPix - 1;
        const int pyk = pc >> lgWout, pxk = pc & (Wout - 1);
        const float mk = maskOut[pc];

        float v0 = lds[j * LDS_ROW + c8 + 0], v1 = lds[j * LDS_ROW + c8 + 1];
        float v2 = lds[j * LDS_ROW + c8 + 2], v3 = lds[j * LDS_ROW + c8 + 3];
        float v4 = lds[j * LDS_ROW + c8 + 4], v5 = lds[j * LDS_ROW + c8 + 5];
        float v6 = lds[j * LDS_ROW + c8 + 6], v7 = lds[j * LDS_ROW + c8 + 7];
        v0 = fmaxf(v0, 0.f) * mk; v1 = fmaxf(v1, 0.f) * mk;
        v2 = fmaxf(v2, 0.f) * mk; v3 = fmaxf(v3, 0.f) * mk;
        v4 = fmaxf(v4, 0.f) * mk; v5 = fmaxf(v5, 0.f) * mk;
        v6 = fmaxf(v6, 0.f) * mk; v7 = fmaxf(v7, 0.f) * mk;
        uint4 st;
        st.x = pack_bf16x2(v0, v1); st.y = pack_bf16x2(v2, v3);
        st.z = pack_bf16x2(v4, v5); st.w = pack_bf16x2(v6, v7);
        if (p < nPix)
            *(uint4*)(out + ((size_t)(pyk + 1) * Woutp + (pxk + 1)) * 32 + nBase + c8) = st;
    }
}

// ---------------------------------------------------------------------------
// Zero the 1-pixel halo ring of a padded (H+2)x(W+2)x32 bf16 map.
// ---------------------------------------------------------------------------
__global__ void zero_border(unsigned short* __restrict__ buf, int H, int W)
{
    const int Hp = H + 2, Wp = W + 2;
    const int nRing = 2 * Wp + 2 * H;
    const uint4 z = make_uint4(0u, 0u, 0u, 0u);
    for (int i = blockIdx.x * blockDim.x + threadIdx.x; i < nRing;
         i += gridDim.x * blockDim.x) {
        int py, px;
        if (i < Wp)            { py = 0;      px = i; }
        else if (i < 2 * Wp)   { py = Hp - 1; px = i - Wp; }
        else { int r = i - 2 * Wp; py = 1 + (r >> 1); px = (r & 1) ? (Wp - 1) : 0; }
        uint4* dst = (uint4*)(buf + ((size_t)py * Wp + px) * 32);
        dst[0] = z; dst[1] = z; dst[2] = z; dst[3] = z;
    }
}

// ---------------------------------------------------------------------------
// 2x2 max-pool of the mask (stride 2, SAME); masks are unpadded.
// ---------------------------------------------------------------------------
__global__ void mask_down(const float* __restrict__ mIn, float* __restrict__ mOut,
                          int Hin, int Win, int Hout, int Wout)
{
    int idx = blockIdx.x * blockDim.x + threadIdx.x;
    if (idx >= Hout * Wout) return;
    int oy = idx / Wout, ox = idx % Wout;
    float v = 0.f;
#pragma unroll
    for (int dy = 0; dy < 2; ++dy)
#pragma unroll
        for (int dx = 0; dx < 2; ++dx) {
            int iy = oy * 2 + dy, ix = ox * 2 + dx;
            if (iy < Hin && ix < Win) v = fmaxf(v, mIn[iy * Win + ix]);
        }
    mOut[idx] = v;
}

// ---------------------------------------------------------------------------
// Masked global-average pool accumulation over a PADDED activation map.
// ---------------------------------------------------------------------------
__global__ void pool_reduce(const unsigned short* __restrict__ y,  // (H+2)*(W+2)*32 bf16
                            const float* __restrict__ mask,        // H*W
                            float* __restrict__ sums,              // 32
                            float* __restrict__ cnt,               // 1
                            int nPix, int W, int lgW)
{
    __shared__ float ls[33];
    if (threadIdx.x < 33) ls[threadIdx.x] = 0.f;
    __syncthreads();

    const int c    = threadIdx.x & 31;
    const int slot = threadIdx.x >> 5;
    const int slots = blockDim.x >> 5;
    const int Wp = W + 2;

    float s = 0.f, mc = 0.f;
    for (int p = blockIdx.x * slots + slot; p < nPix; p += gridDim.x * slots) {
        int py = p >> lgW, px = p & (W - 1);
        s += bf16_to_f32(y[((size_t)(py + 1) * Wp + (px + 1)) * 32 + c]);
        if (c == 0) mc += mask[p];
    }
    atomicAdd(&ls[c], s);
    if (c == 0) atomicAdd(&ls[32], mc);
    __syncthreads();
    if (threadIdx.x < 32) atomicAdd(&sums[threadIdx.x], ls[threadIdx.x]);
    if (threadIdx.x == 32) atomicAdd(cnt, ls[32]);
}

// ---------------------------------------------------------------------------
// Weight prep: f32 -> bf16, layer-1 weights padded K 25 -> 32 (zeros).
// ---------------------------------------------------------------------------
__global__ void prep_weights(const float* __restrict__ w1,   // 25*32
                             const float* __restrict__ ws,   // 13*288*32
                             unsigned short* __restrict__ B0,
                             unsigned short* __restrict__ Bs,
                             int nWs)
{
    int i = blockIdx.x * blockDim.x + threadIdx.x;
    if (i < 32 * 32) {
        int k = i >> 5;
        B0[i] = (k < 25) ? f32_to_bf16(w1[k * 32 + (i & 31)]) : (unsigned short)0;
    }
    if (i < nWs) Bs[i] = f32_to_bf16(ws[i]);
}

__global__ void init_pools(float* __restrict__ sums, float* __restrict__ cnt)
{
    int i = threadIdx.x;
    if (i < 14 * 32) sums[i] = 0.f;
    if (i < 14) cnt[i] = 0.f;
}

// ---------------------------------------------------------------------------
// Final MLP: feat(448) -> relu(448x256) -> 256x128. Single block.
// ---------------------------------------------------------------------------
__global__ void mlp_head(const float* __restrict__ sums, const float* __restrict__ cnt,
                         const float* __restrict__ wm1, const float* __restrict__ bm1,
                         const float* __restrict__ wm2, const float* __restrict__ bm2,
                         float* __restrict__ outp)
{
    __shared__ float feat[448];
    __shared__ float h[256];
    const int t = threadIdx.x;   // 256 threads

    for (int i = t; i < 448; i += 256) {
        float c = cnt[i >> 5];
        c = c > 1.f ? c : 1.f;
        feat[i] = sums[i] / c;
    }
    __syncthreads();

    {
        float a = bm1[t];
        for (int k = 0; k < 448; ++k) a = fmaf(feat[k], wm1[k * 256 + t], a);
        h[t] = a > 0.f ? a : 0.f;
    }
    __syncthreads();

    if (t < 128) {
        float a = bm2[t];
        for (int k = 0; k < 256; ++k) a = fmaf(h[k], wm2[k * 128 + t], a);
        outp[t] = a;
    }
}

// ---------------------------------------------------------------------------
// Orchestration
// ---------------------------------------------------------------------------
static inline int conv_blocks(int nPix) {
    long nTiles = (nPix + 15) / 16;
    long waves  = nTiles * 2;            // two N-halves
    long blocks = (waves + 3) / 4;       // 4 waves (128 threads) per block
    if (blocks > 4096) blocks = 4096;
    if (blocks < 1) blocks = 1;
    return (int)blocks;
}

extern "C" void kernel_launch(void* const* d_in, const int* in_sizes, int n_in,
                              void* d_out, int out_size, void* d_ws, size_t ws_size,
                              hipStream_t stream)
{
    const float* x    = (const float*)d_in[0];   // 1,2048,2048,1
    const float* mask0= (const float*)d_in[1];   // 1,2048,2048
    const float* w1   = (const float*)d_in[2];   // 5,5,1,32
    const float* ws   = (const float*)d_in[3];   // 13,3,3,32,32
    const float* wm1  = (const float*)d_in[4];   // 448,256
    const float* bm1  = (const float*)d_in[5];   // 256
    const float* wm2  = (const float*)d_in[6];   // 256,128
    const float* bm2  = (const float*)d_in[7];   // 128
    float* outp = (float*)d_out;                 // 128

    char* base = (char*)d_ws;

    // ---- workspace layout (byte offsets, 256-aligned) ----
    const size_t offSums = 0;                          // 14*32 f32
    const size_t offCnt  = 2048;                       // 14 f32
    const size_t offB0   = 4096;                       // 32*32 bf16
    const size_t offBs   = 8192;                       // 13*288*32 bf16
    size_t cur = offBs + (size_t)13 * 288 * 32 * 2;
    cur = (cur + 255) & ~(size_t)255;

    int dims[14], lgs[14];
    dims[0] = 2048;
    for (int l = 1; l < 14; ++l) { dims[l] = dims[l - 1] / 2; if (dims[l] < 1) dims[l] = 1; }
    for (int l = 0; l < 14; ++l) { int g = 0; while ((1 << g) < dims[l]) ++g; lgs[l] = g; }

    size_t maskOff[14];
    maskOff[0] = (size_t)-1;
    for (int l = 1; l < 14; ++l) {
        maskOff[l] = cur;
        cur += (size_t)dims[l] * dims[l] * sizeof(float);
        cur = (cur + 255) & ~(size_t)255;
    }

    // padded bf16 input copy: (2048+4)^2 * 2 bytes
    const size_t offXP = (cur + 255) & ~(size_t)255;
    cur = offXP + (size_t)2052 * 2052 * 2;

    // ping-pong PADDED activations (bf16, C=32): (d+2)^2 * 32 * 2 bytes
    const size_t offYA = (cur + 255) & ~(size_t)255;           // levels 0,2,4,...
    const size_t offYB = offYA + (size_t)2050 * 2050 * 32 * 2; // levels 1,3,...

    float* sums = (float*)(base + offSums);
    float* cnt  = (float*)(base + offCnt);
    unsigned short* B0 = (unsigned short*)(base + offB0);
    unsigned short* Bs = (unsigned short*)(base + offBs);
    unsigned short* xpad = (unsigned short*)(base + offXP);
    unsigned short* yA = (unsigned short*)(base + offYA);
    unsigned short* yB = (unsigned short*)(base + offYB);

    const float* mPtr[14];
    mPtr[0] = mask0;
    for (int l = 1; l < 14; ++l) mPtr[l] = (const float*)(base + maskOff[l]);

    // 1) zero pool accumulators
    init_pools<<<1, 512, 0, stream>>>(sums, cnt);

    // 2) bf16 weight prep + padded bf16 input copy
    {
        const int nWs = 13 * 288 * 32;
        int blocks = (nWs + 255) / 256;
        prep_weights<<<blocks, 256, 0, stream>>>(w1, ws, B0, Bs, nWs);
        int nP = 2052 * 2052;
        int pblocks = (nP + 255) / 256; if (pblocks > 8192) pblocks = 8192;
        pad_input<<<pblocks, 256, 0, stream>>>(x, xpad, 2048, 2048);
    }

    // 3) mask pyramid
    for (int l = 1; l < 14; ++l) {
        int nOut = dims[l] * dims[l];
        int blocks = (nOut + 255) / 256;
        mask_down<<<blocks, 256, 0, stream>>>(mPtr[l - 1], (float*)(base + maskOff[l]),
                                              dims[l - 1], dims[l - 1], dims[l], dims[l]);
    }

    // 4) layer 1 (5x5, stride 1) -> padded yA, pool level 0
    {
        int nPix = dims[0] * dims[0];
        int ring = 2 * (dims[0] + 2) + 2 * dims[0];
        zero_border<<<(ring + 255) / 256, 256, 0, stream>>>(yA, dims[0], dims[0]);
        conv5x5_wmma<<<conv_blocks(nPix), 128, 0, stream>>>(xpad, yA, mPtr[0], B0,
                                                            dims[0], dims[0], lgs[0]);
        int pb = (nPix + 7) / 8; if (pb > 2048) pb = 2048; if (pb < 1) pb = 1;
        pool_reduce<<<pb, 256, 0, stream>>>(yA, mPtr[0], sums + 0, cnt + 0,
                                            nPix, dims[0], lgs[0]);
    }

    // 5) 13 stride-2 layers (ping-pong), pool after each
    for (int l = 1; l < 14; ++l) {
        const unsigned short* in = (l & 1) ? yA : yB;
        unsigned short* out      = (l & 1) ? yB : yA;
        int nPix = dims[l] * dims[l];
        int ring = 2 * (dims[l] + 2) + 2 * dims[l];
        zero_border<<<(ring + 255) / 256, 256, 0, stream>>>(out, dims[l], dims[l]);
        conv3x3s2_wmma<<<conv_blocks(nPix), 128, 0, stream>>>(
            in, out, mPtr[l], Bs + (size_t)(l - 1) * 288 * 32,
            dims[l - 1], dims[l], dims[l], lgs[l]);
        int pb = (nPix + 7) / 8; if (pb > 2048) pb = 2048; if (pb < 1) pb = 1;
        pool_reduce<<<pb, 256, 0, stream>>>(out, mPtr[l], sums + l * 32, cnt + l,
                                            nPix, dims[l], lgs[l]);
    }

    // 6) MLP head
    mlp_head<<<1, 256, 0, stream>>>(sums, cnt, wm1, bm1, wm2, bm2, outp);

    (void)in_sizes; (void)n_in; (void)out_size; (void)ws_size;
}
